// HierarchicalRNNDecoder_86990267613622
// MI455X (gfx1250) — matrix operand order repaired and checked by reading, not compiled, not run on hardware
//
#include <hip/hip_runtime.h>
#include <hip/hip_bf16.h>
#include <math.h>

// ---------------------------------------------------------------------------
// HierarchicalRNNDecoder for MI455X (gfx1250, wave32, WMMA).
//
// Shapes: B=128, Z=128, H=256, TOP_REP=32, CHUNK=64, T2=2048.
// Level-2 GRU (2048 sequential steps of h[16x256] @ Whh2^T[256x768] per
// batch-tile) dominates -> persistent per-batch-tile workgroups, h in
// registers + bf16 LDS mirror, Whh in L2-resident bf16 packed layout,
// v_wmma_f32_16x16x32_bf16 with f32 accumulation, 2 barriers/step.
// ---------------------------------------------------------------------------

typedef __attribute__((ext_vector_type(16))) __bf16 v16bf;
typedef __attribute__((ext_vector_type(8)))  float v8f;

union FragA { v16bf v; unsigned int u[8]; };

__device__ __forceinline__ unsigned short f2bf(float f) {
    union { float f; unsigned int u; } x; x.f = f;
    unsigned int r = x.u + 0x7fffu + ((x.u >> 16) & 1u);   // RNE
    return (unsigned short)(r >> 16);
}
__device__ __forceinline__ float sigm(float x) { return 1.0f / (1.0f + __expf(-x)); }
__device__ __forceinline__ float eluf(float x) { return x > 0.0f ? x : (__expf(x) - 1.0f); }

// ---------------------------------------------------------------------------
// Pack Whh (768x256 f32, row = gate output, col = hidden k) into bf16 WMMA
// B-fragment order: Wp[((nt*8+kf)*32+lane)*16 + i]
//   lane<16 : column n = nt*16+lane, K = kf*32 + 0..15
//   lane>=16: column n = nt*16+lane-16, K = kf*32 + 16..31
// so each lane reads one contiguous 32B v16bf per fragment.
// ---------------------------------------------------------------------------
__global__ void pack_whh_kernel(const float* __restrict__ W,
                                unsigned short* __restrict__ Wp) {
    int idx = blockIdx.x * blockDim.x + threadIdx.x;           // 48*8*32*16
    if (idx >= 48 * 8 * 32 * 16) return;
    int i    = idx & 15;
    int lane = (idx >> 4) & 31;
    int kf   = (idx >> 9) & 7;
    int nt   = idx >> 12;
    int n = nt * 16 + (lane & 15);
    int k = kf * 32 + ((lane >> 4) << 4) + i;
    Wp[idx] = f2bf(W[n * 256 + k]);
}

// h0 = z @ W_z2h0^T + b_z2h0 ; inp = z @ W_z2h^T + b_z2h
__global__ void init_vec_kernel(const float* __restrict__ z,
                                const float* __restrict__ Wz,  const float* __restrict__ bz,
                                const float* __restrict__ Wz0, const float* __restrict__ bz0,
                                float* __restrict__ inp, float* __restrict__ h0) {
    int idx = blockIdx.x * blockDim.x + threadIdx.x;           // 2*128*256
    if (idx >= 2 * 128 * 256) return;
    int o = idx & 255, b = (idx >> 8) & 127, which = idx >> 15;
    const float* W  = which ? Wz0 : Wz;
    const float* bi = which ? bz0 : bz;
    float s = bi[o];
    const float* zr = z + b * 128;
    const float* wr = W + o * 128;
    for (int k = 0; k < 128; ++k) s += zr[k] * wr[k];
    (which ? h0 : inp)[b * 256 + o] = s;
}

// c1[b][g] = inp[b]@Wih1[g]^T + bih1[g] + (g<512 ? bhh1[g] : 0)   (bhh folded
// into r,z pre-activations; n's input part kept separate from hidden part)
__global__ void c1_prep_kernel(const float* __restrict__ inp,
                               const float* __restrict__ Wih1,
                               const float* __restrict__ bih1,
                               const float* __restrict__ bhh1,
                               float* __restrict__ c1) {
    int idx = blockIdx.x * blockDim.x + threadIdx.x;           // 128*768
    if (idx >= 128 * 768) return;
    int g = idx % 768, b = idx / 768;
    float s = bih1[g] + (g < 512 ? bhh1[g] : 0.0f);
    const float* x = inp + b * 256;
    const float* w = Wih1 + g * 256;
    for (int k = 0; k < 256; ++k) s += x[k] * w[k];
    c1[idx] = s;
}

// out1[t][b][o] = elu(top)[t][b] @ W_out1[o]^T + b_out1[o]; top_e already elu'd
__global__ void out1_gemm_kernel(const float* __restrict__ top_e,
                                 const float* __restrict__ Wo,
                                 const float* __restrict__ bo,
                                 float* __restrict__ out1) {
    int idx = blockIdx.x * blockDim.x + threadIdx.x;           // 32*128*256
    if (idx >= 32 * 128 * 256) return;
    int o = idx & 255;
    int tb = idx >> 8;                                         // t*128 + b
    float s = bo[o];
    const float* x = top_e + tb * 256;
    const float* w = Wo + o * 256;
    for (int k = 0; k < 256; ++k) s += x[k] * w[k];
    out1[idx] = s;
}

// c2[t][b][g] = out1[t][b]@Wih2[g]^T + bih2[g] + (g<512 ? bhh2[g] : 0)
__global__ void c2_prep_kernel(const float* __restrict__ out1,
                               const float* __restrict__ Wih2,
                               const float* __restrict__ bih2,
                               const float* __restrict__ bhh2,
                               float* __restrict__ c2) {
    int idx = blockIdx.x * blockDim.x + threadIdx.x;           // 32*128*768
    if (idx >= 32 * 128 * 768) return;
    int g = idx % 768;
    int tb = idx / 768;
    float s = bih2[g] + (g < 512 ? bhh2[g] : 0.0f);
    const float* x = out1 + tb * 256;
    const float* w = Wih2 + g * 256;
    for (int k = 0; k < 256; ++k) s += x[k] * w[k];
    c2[idx] = s;
}

// ---------------------------------------------------------------------------
// Level-1 GRU: 32 steps, batch tile of 16 rows per workgroup, 8 waves.
// Wave w owns hidden-column tiles {2w, 2w+1}; accumulators:
//   acc[g*2+u] -> gate g (0=r,1=z,2=n), column tile 2w+u.
// Writes elu(h_t) to top_e[t][b][j].
// ---------------------------------------------------------------------------
__global__ __launch_bounds__(256) void gru_level1_kernel(
    const float* __restrict__ h0, const float* __restrict__ c1,
    const float* __restrict__ bhh1, const unsigned short* __restrict__ Wp,
    float* __restrict__ top_e) {
    __shared__ unsigned short h_lds[16 * 256];                 // bf16 mirror of h
    const int tid  = threadIdx.x;
    const int lane = tid & 31;
    const int wv   = tid >> 5;                                 // wave 0..7
    const int half = lane >> 4;                                // 0/1
    const int lm   = lane & 15;
    const int m0   = blockIdx.x * 16;                          // batch tile base

    float cr[2][8], cz[2][8], gn[2][8], hreg[2][8], bn[2];
    #pragma unroll
    for (int u = 0; u < 2; ++u) {
        int j = (2 * wv + u) * 16 + lm;
        bn[u] = bhh1[512 + j];
        #pragma unroll
        for (int v = 0; v < 8; ++v) {
            int b = m0 + v + 8 * half;
            cr[u][v] = c1[b * 768 + j];
            cz[u][v] = c1[b * 768 + 256 + j];
            gn[u][v] = c1[b * 768 + 512 + j];
            float h = h0[b * 256 + j];
            hreg[u][v] = h;
            h_lds[(v + 8 * half) * 256 + j] = f2bf(h);
        }
    }
    __syncthreads();

    for (int t = 0; t < 32; ++t) {
        v8f acc[6] = {};
        #pragma unroll
        for (int kf = 0; kf < 8; ++kf) {
            FragA a;
            const int k0 = kf * 32;
            #pragma unroll
            for (int q = 0; q < 8; ++q) {   // A layout: K = k0 + 0..15 (+16 hi-half)
                int kb = k0 + half * 16 + (q << 1);
                a.u[q] = *(const unsigned int*)&h_lds[lm * 256 + kb];
            }
            #pragma unroll
            for (int g = 0; g < 3; ++g) {
                #pragma unroll
                for (int u = 0; u < 2; ++u) {
                    int nt = g * 16 + 2 * wv + u;
                    FragA bfr;
                    bfr.v = *(const v16bf*)(Wp + ((size_t)(nt * 8 + kf) * 32 + lane) * 16);
                    acc[g * 2 + u] = __builtin_amdgcn_wmma_f32_16x16x32_bf16(
                        false, a.v, false, bfr.v, (short)0, acc[g * 2 + u], false, false);
                }
            }
        }
        __syncthreads();   // all h_lds reads done before rewriting h
        #pragma unroll
        for (int u = 0; u < 2; ++u) {
            int j = (2 * wv + u) * 16 + lm;
            #pragma unroll
            for (int v = 0; v < 8; ++v) {
                int b = m0 + v + 8 * half;
                float r  = sigm(acc[0 + u][v] + cr[u][v]);
                float zt = sigm(acc[2 + u][v] + cz[u][v]);
                float n  = tanhf(gn[u][v] + r * (acc[4 + u][v] + bn[u]));
                float h  = (1.0f - zt) * n + zt * hreg[u][v];
                hreg[u][v] = h;
                h_lds[(v + 8 * half) * 256 + j] = f2bf(h);
                top_e[((size_t)t * 128 + b) * 256 + j] = eluf(h);
            }
        }
        __syncthreads();
    }
}

// ---------------------------------------------------------------------------
// Level-2 GRU: 2048 steps; gi constants reload every 64-step chunk; output
// scalar per (b,t) reduced via LDS float atomics: out[b][t]=elu(h)·W_out2+b.
// ---------------------------------------------------------------------------
__global__ __launch_bounds__(256) void gru_level2_kernel(
    const float* __restrict__ h0sec,        // out1_all[0] : (128,256)
    const float* __restrict__ c2,           // (32,128,768)
    const float* __restrict__ bhh2,
    const float* __restrict__ W_out2,       // (256)
    const float* __restrict__ b_out2,       // (1)
    const unsigned short* __restrict__ Wp,
    float* __restrict__ out) {              // (128, 2048)
    __shared__ unsigned short h_lds[16 * 256];
    __shared__ float out_acc[16];
    const int tid  = threadIdx.x;
    const int lane = tid & 31;
    const int wv   = tid >> 5;
    const int half = lane >> 4;
    const int lm   = lane & 15;
    const int m0   = blockIdx.x * 16;

    float cr[2][8], cz[2][8], gn[2][8], hreg[2][8], bn[2], w2[2];
    #pragma unroll
    for (int u = 0; u < 2; ++u) {
        int j = (2 * wv + u) * 16 + lm;
        bn[u] = bhh2[512 + j];
        w2[u] = W_out2[j];
        #pragma unroll
        for (int v = 0; v < 8; ++v) {
            int b = m0 + v + 8 * half;
            float h = h0sec[b * 256 + j];
            hreg[u][v] = h;
            h_lds[(v + 8 * half) * 256 + j] = f2bf(h);
        }
    }
    if (tid < 16) out_acc[tid] = 0.0f;
    const float bout = b_out2[0];
    __syncthreads();

    for (int t = 0; t < 2048; ++t) {
        if ((t & 63) == 0) {                                   // new chunk constants
            const float* cc = c2 + (size_t)(t >> 6) * 128 * 768;
            #pragma unroll
            for (int u = 0; u < 2; ++u) {
                int j = (2 * wv + u) * 16 + lm;
                #pragma unroll
                for (int v = 0; v < 8; ++v) {
                    int b = m0 + v + 8 * half;
                    cr[u][v] = cc[b * 768 + j];
                    cz[u][v] = cc[b * 768 + 256 + j];
                    gn[u][v] = cc[b * 768 + 512 + j];
                }
            }
        }
        v8f acc[6] = {};
        #pragma unroll
        for (int kf = 0; kf < 8; ++kf) {
            FragA a;
            const int k0 = kf * 32;
            #pragma unroll
            for (int q = 0; q < 8; ++q) {
                int kb = k0 + half * 16 + (q << 1);
                a.u[q] = *(const unsigned int*)&h_lds[lm * 256 + kb];
            }
            #pragma unroll
            for (int g = 0; g < 3; ++g) {
                #pragma unroll
                for (int u = 0; u < 2; ++u) {
                    int nt = g * 16 + 2 * wv + u;
                    FragA bfr;
                    bfr.v = *(const v16bf*)(Wp + ((size_t)(nt * 8 + kf) * 32 + lane) * 16);
                    acc[g * 2 + u] = __builtin_amdgcn_wmma_f32_16x16x32_bf16(
                        false, a.v, false, bfr.v, (short)0, acc[g * 2 + u], false, false);
                }
            }
        }
        __syncthreads();   // h_lds reads complete; out_acc(t-1) retired
        #pragma unroll
        for (int v = 0; v < 8; ++v) {
            float p = 0.0f;
            #pragma unroll
            for (int u = 0; u < 2; ++u) {
                int j = (2 * wv + u) * 16 + lm;
                float r  = sigm(acc[0 + u][v] + cr[u][v]);
                float zt = sigm(acc[2 + u][v] + cz[u][v]);
                float n  = tanhf(gn[u][v] + r * (acc[4 + u][v] + bn[u]));
                float h  = (1.0f - zt) * n + zt * hreg[u][v];
                hreg[u][v] = h;
                h_lds[(v + 8 * half) * 256 + j] = f2bf(h);
                p += w2[u] * eluf(h);
            }
            atomicAdd(&out_acc[v + 8 * half], p);              // ds_add_f32
        }
        __syncthreads();   // h_lds + out_acc of step t visible
        if (tid < 16) {
            out[(size_t)(m0 + tid) * 2048 + t] = out_acc[tid] + bout;
            out_acc[tid] = 0.0f;   // next step's atomics are gated by next barrier
        }
    }
}

// ---------------------------------------------------------------------------
// Host orchestration. Workspace layout (bytes):
// ---------------------------------------------------------------------------
static const size_t OFF_WP1  = 0;                        // 384 KB bf16 packed Whh1
static const size_t OFF_WP2  = OFF_WP1 + 393216;         // 384 KB bf16 packed Whh2
static const size_t OFF_H0   = OFF_WP2 + 393216;         // 128x256 f32
static const size_t OFF_INP  = OFF_H0  + 131072;         // 128x256 f32
static const size_t OFF_C1   = OFF_INP + 131072;         // 128x768 f32
static const size_t OFF_TOPE = OFF_C1  + 393216;         // 32x128x256 f32 (elu'd)
static const size_t OFF_OUT1 = OFF_TOPE + 4194304;       // 32x128x256 f32
static const size_t OFF_C2   = OFF_OUT1 + 4194304;       // 32x128x768 f32
// total ~22.4 MB

extern "C" void kernel_launch(void* const* d_in, const int* in_sizes, int n_in,
                              void* d_out, int out_size, void* d_ws, size_t ws_size,
                              hipStream_t stream) {
    (void)in_sizes; (void)n_in; (void)out_size; (void)ws_size;
    const float* z      = (const float*)d_in[0];
    const float* W_z2h  = (const float*)d_in[1];
    const float* b_z2h  = (const float*)d_in[2];
    const float* W_z2h0 = (const float*)d_in[3];
    const float* b_z2h0 = (const float*)d_in[4];
    const float* Wih1   = (const float*)d_in[5];
    const float* Whh1   = (const float*)d_in[6];
    const float* bih1   = (const float*)d_in[7];
    const float* bhh1   = (const float*)d_in[8];
    const float* W_out1 = (const float*)d_in[9];
    const float* b_out1 = (const float*)d_in[10];
    const float* Wih2   = (const float*)d_in[11];
    const float* Whh2   = (const float*)d_in[12];
    const float* bih2   = (const float*)d_in[13];
    const float* bhh2   = (const float*)d_in[14];
    const float* W_out2 = (const float*)d_in[15];
    const float* b_out2 = (const float*)d_in[16];

    char* ws = (char*)d_ws;
    unsigned short* Wp1 = (unsigned short*)(ws + OFF_WP1);
    unsigned short* Wp2 = (unsigned short*)(ws + OFF_WP2);
    float* h0    = (float*)(ws + OFF_H0);
    float* inp   = (float*)(ws + OFF_INP);
    float* c1    = (float*)(ws + OFF_C1);
    float* top_e = (float*)(ws + OFF_TOPE);
    float* out1  = (float*)(ws + OFF_OUT1);
    float* c2    = (float*)(ws + OFF_C2);

    pack_whh_kernel<<<768, 256, 0, stream>>>(Whh1, Wp1);
    pack_whh_kernel<<<768, 256, 0, stream>>>(Whh2, Wp2);
    init_vec_kernel<<<256, 256, 0, stream>>>(z, W_z2h, b_z2h, W_z2h0, b_z2h0, inp, h0);
    c1_prep_kernel<<<384, 256, 0, stream>>>(inp, Wih1, bih1, bhh1, c1);
    gru_level1_kernel<<<8, 256, 0, stream>>>(h0, c1, bhh1, Wp1, top_e);
    out1_gemm_kernel<<<4096, 256, 0, stream>>>(top_e, W_out1, b_out1, out1);
    c2_prep_kernel<<<12288, 256, 0, stream>>>(out1, Wih2, bih2, bhh2, c2);
    gru_level2_kernel<<<8, 256, 0, stream>>>(out1 /* out1_all[0] */, c2, bhh2,
                                             W_out2, b_out2, Wp2, (float*)d_out);
}
